// GCNNet_62225486184658
// MI455X (gfx1250) — matrix operand order repaired and checked by reading, not compiled, not run on hardware
//
#include <hip/hip_runtime.h>

#define NN     50000
#define EE     1600000
#define N_PAD  50048            // multiple of 64
#define MG     (N_PAD / 64)     // 782 m-groups of 64 rows

typedef __attribute__((ext_vector_type(16))) __bf16 v16bf;
typedef __attribute__((ext_vector_type(8)))  __bf16 v8bf;
typedef __attribute__((ext_vector_type(8)))  float  v8f;

// ---------------------------------------------------------------------------
// WMMA GEMM: C[N_PAD, ldc] (f32) = A[N_PAD, 256] (bf16, lda=256) @ packed-B.
// One wave = 64 rows x 32 cols (2 N-tiles): each A fragment feeds 2 WMMAs.
// Branchless inner loop: for odd nt_count the second B fragment aliases the
// first (wasted but harmless); guard only applies to the final store.
// ---------------------------------------------------------------------------
__global__ __launch_bounds__(128)
void gemm_bf16_wmma(const __bf16* __restrict__ A, const __bf16* __restrict__ Bp,
                    float* __restrict__ C, int ldc, int nt_count, int kt_count,
                    int n_groups)
{
    int wave = blockIdx.x * 4 + (threadIdx.x >> 5);
    if (wave >= n_groups) return;                 // wave-uniform: EXEC all-1s inside
    int lane = threadIdx.x & 31;
    int mg  = wave % MG;
    int np  = wave / MG;                          // N-tile pair index
    int nt0 = np * 2;
    bool has2 = (nt0 + 1) < nt_count;             // wave-uniform
    size_t b1off = has2 ? (size_t)(32 * 16) : 0;  // element delta to 2nd fragment
    int m0 = mg * 64;
    int r16 = lane & 15, grp = lane >> 4;

    v8f acc0[4], acc1[4];
#pragma unroll
    for (int t = 0; t < 4; ++t)
#pragma unroll
        for (int j = 0; j < 8; ++j) { acc0[t][j] = 0.0f; acc1[t][j] = 0.0f; }

    for (int kt = 0; kt < kt_count; ++kt) {
        // B fragments: pre-packed in exact lane order, one 32B load each
        const __bf16* bbase = Bp + (((size_t)kt * nt_count + nt0) * 32 + lane) * 16;
        v16bf b0 = *(const v16bf*)(bbase);
        v16bf b1 = *(const v16bf*)(bbase + b1off);
        // A fragment base for this lane: row m0+r16, K = kt*32 + 8*grp (+16)
        const __bf16* abase = A + ((size_t)(m0 + r16)) * 256 + kt * 32 + 8 * grp;
#pragma unroll
        for (int t = 0; t < 4; ++t) {
            const __bf16* p = abase + (size_t)t * 16 * 256;
            v8bf lo = *(const v8bf*)(p);        // K = kbase .. kbase+7
            v8bf hi = *(const v8bf*)(p + 16);   // K = kbase+16 .. kbase+23
            v16bf af;
#pragma unroll
            for (int j = 0; j < 8; ++j) { af[j] = lo[j]; af[j + 8] = hi[j]; }
            acc0[t] = __builtin_amdgcn_wmma_f32_16x16x32_bf16(
                false, af, false, b0, (short)0, acc0[t], false, false);
            acc1[t] = __builtin_amdgcn_wmma_f32_16x16x32_bf16(
                false, af, false, b1, (short)0, acc1[t], false, false);
        }
    }

    int n0 = nt0 * 16;
#pragma unroll
    for (int t = 0; t < 4; ++t) {
        float* c = C + ((size_t)(m0 + t * 16 + grp * 8)) * ldc + n0 + r16;
#pragma unroll
        for (int j = 0; j < 8; ++j) c[(size_t)j * ldc] = acc0[t][j];
    }
    if (has2) {
#pragma unroll
        for (int t = 0; t < 4; ++t) {
            float* c = C + ((size_t)(m0 + t * 16 + grp * 8)) * ldc + n0 + 16 + r16;
#pragma unroll
            for (int j = 0; j < 8; ++j) c[(size_t)j * ldc] = acc1[t][j];
        }
    }
}

// ---------------------------------------------------------------------------
// Pack fp32 W[fi, fo] (row-major) into WMMA B-fragment order, bf16.
// Fragment layout (32x16 K x N tile): lane = 16*grpB + n,
//   element j = W[kt*32 + 8*grpB + (j&7) + 16*(j>>3)][nt*16 + n]
// ---------------------------------------------------------------------------
__global__ void pack_w(const float* __restrict__ W, __bf16* __restrict__ out,
                       int fi, int fo_real, int fo_pad)
{
    int idx = blockIdx.x * blockDim.x + threadIdx.x;
    int total = fi * fo_pad;
    if (idx >= total) return;
    int j    = idx & 15;
    int lane = (idx >> 4) & 31;
    int tile = idx >> 9;                 // kt * nt_count + nt
    int nt_count = fo_pad >> 4;
    int nt = tile % nt_count;
    int kt = tile / nt_count;
    int grp = lane >> 4, nn = lane & 15;
    int k = kt * 32 + 8 * grp + (j & 7) + 16 * (j >> 3);
    int n = nt * 16 + nn;
    float v = (n < fo_real) ? W[(size_t)k * fo_real + n] : 0.0f;
    out[idx] = (__bf16)v;
}

// ---------------------------------------------------------------------------
// Graph normalization precompute
// ---------------------------------------------------------------------------
__global__ void deg_init(float* deg)
{
    int n = blockIdx.x * blockDim.x + threadIdx.x;
    if (n < NN) deg[n] = 1.0f;                     // self-loop weight
}
__global__ void deg_acc(const int* __restrict__ col, const float* __restrict__ ew,
                        float* deg)
{
    int e = blockIdx.x * blockDim.x + threadIdx.x;
    if (e < EE) atomicAdd(&deg[col[e]], ew[e]);
}
__global__ void dinv_k(const float* __restrict__ deg, float* dinv, float* selfn)
{
    int n = blockIdx.x * blockDim.x + threadIdx.x;
    if (n >= NN) return;
    float d = deg[n];
    float r = d > 0.0f ? rsqrtf(d) : 0.0f;
    dinv[n] = r;
    selfn[n] = r * r;
}
__global__ void norm_k(const int* __restrict__ row, const int* __restrict__ col,
                       const float* __restrict__ ew, const float* __restrict__ dinv,
                       float* __restrict__ norm)
{
    int e = blockIdx.x * blockDim.x + threadIdx.x;
    if (e < EE) norm[e] = dinv[row[e]] * ew[e] * dinv[col[e]];
}

// ---------------------------------------------------------------------------
// BN on input -> bf16 (lda = 256)
// ---------------------------------------------------------------------------
__global__ void bn_input(const float* __restrict__ x,
                         const float* __restrict__ g, const float* __restrict__ b,
                         const float* __restrict__ m, const float* __restrict__ v,
                         __bf16* __restrict__ h)
{
    int idx = blockIdx.x * blockDim.x + threadIdx.x;
    if (idx >= NN * 256) return;
    int f = idx & 255;
    float y = g[f] * (x[idx] - m[f]) * rsqrtf(v[f] + 1e-5f) + b[f];
    h[idx] = (__bf16)y;
}

// ---------------------------------------------------------------------------
// Edge scatter: accum[col] += norm * hW[row].
// thread = (edge, float4 chunk): lanes cover contiguous features -> coalesced
// gathers and coalesced atomic bursts. fo = 4 << fo4_shift.
// ---------------------------------------------------------------------------
__global__ void scatter_edges(const float* __restrict__ hW,
                              const int* __restrict__ row, const int* __restrict__ col,
                              const float* __restrict__ norm,
                              float* __restrict__ accum, int fo4_shift)
{
    int t = blockIdx.x * blockDim.x + threadIdx.x;
    int e = t >> fo4_shift;
    if (e >= EE) return;
    int c4 = t & ((1 << fo4_shift) - 1);
    int r = row[e], c = col[e];
    float w = norm[e];
    const float4 v = *(const float4*)(hW + (((size_t)r << fo4_shift) + c4) * 4);
    float* dst = accum + (((size_t)c << fo4_shift) + c4) * 4;
    atomicAdd(dst + 0, w * v.x);
    atomicAdd(dst + 1, w * v.y);
    atomicAdd(dst + 2, w * v.z);
    atomicAdd(dst + 3, w * v.w);
}

// ---------------------------------------------------------------------------
// finalize: h_next = relu(BN(accum + selfnorm*hW + bias))   -> bf16, lda 256
// fo = 1 << fo_shift
// ---------------------------------------------------------------------------
__global__ void finalize_conv(const float* __restrict__ accum, const float* __restrict__ hW,
                              const float* __restrict__ selfn, const float* __restrict__ bias,
                              const float* __restrict__ g, const float* __restrict__ bt,
                              const float* __restrict__ mn, const float* __restrict__ vr,
                              __bf16* __restrict__ hnext, int fo_shift)
{
    int idx = blockIdx.x * blockDim.x + threadIdx.x;
    if (idx >= (NN << fo_shift)) return;
    int n = idx >> fo_shift;
    int f = idx & ((1 << fo_shift) - 1);
    float y = accum[idx] + selfn[n] * hW[idx] + bias[f];
    float z = g[f] * (y - mn[f]) * rsqrtf(vr[f] + 1e-5f) + bt[f];
    z = fmaxf(z, 0.0f);
    hnext[(size_t)n * 256 + f] = (__bf16)z;
}

// lin0 epilogue: h = relu(BN9(hW + bias)) -> bf16 lda 256  (fo = 64)
__global__ void finalize_lin(const float* __restrict__ hW, const float* __restrict__ bias,
                             const float* __restrict__ g, const float* __restrict__ bt,
                             const float* __restrict__ mn, const float* __restrict__ vr,
                             __bf16* __restrict__ hnext)
{
    int idx = blockIdx.x * blockDim.x + threadIdx.x;
    if (idx >= NN * 64) return;
    int n = idx >> 6;
    int f = idx & 63;
    float y = hW[idx] + bias[f];
    float z = g[f] * (y - mn[f]) * rsqrtf(vr[f] + 1e-5f) + bt[f];
    z = fmaxf(z, 0.0f);
    hnext[(size_t)n * 256 + f] = (__bf16)z;
}

// lin1 epilogue: out[n,10] = hW[n,16 pad] + bias  (fp32 output)
__global__ void write_out(const float* __restrict__ hW, const float* __restrict__ bias,
                          float* __restrict__ out)
{
    int idx = blockIdx.x * blockDim.x + threadIdx.x;
    if (idx >= NN * 10) return;
    int n = idx / 10, c = idx - n * 10;
    out[idx] = hW[(size_t)n * 16 + c] + bias[c];
}

// ---------------------------------------------------------------------------
extern "C" void kernel_launch(void* const* d_in, const int* in_sizes, int n_in,
                              void* d_out, int out_size, void* d_ws, size_t ws_size,
                              hipStream_t stream)
{
    (void)in_sizes; (void)n_in; (void)out_size; (void)ws_size;

    const float* x  = (const float*)d_in[0];
    const int*   ei = (const int*)d_in[1];
    const int*   rowp = ei;
    const int*   colp = ei + EE;
    const float* ew = (const float*)d_in[2];
    const float* convW[8]; const float* convB[8];
    for (int i = 0; i < 8; ++i) { convW[i] = (const float*)d_in[3 + 2*i]; convB[i] = (const float*)d_in[4 + 2*i]; }
    const float* bnG[10]; const float* bnB[10]; const float* bnM[10]; const float* bnV[10];
    for (int j = 0; j < 10; ++j) {
        bnG[j] = (const float*)d_in[19 + 4*j]; bnB[j] = (const float*)d_in[20 + 4*j];
        bnM[j] = (const float*)d_in[21 + 4*j]; bnV[j] = (const float*)d_in[22 + 4*j];
    }
    const float* l0W = (const float*)d_in[59]; const float* l0b = (const float*)d_in[60];
    const float* l1W = (const float*)d_in[61]; const float* l1b = (const float*)d_in[62];
    float* out = (float*)d_out;

    // ---- workspace carve-up (256B aligned blocks) ----
    char* ws = (char*)d_ws; size_t off = 0;
    auto take = [&](size_t bytes) -> char* {
        char* p = ws + off; off = (off + bytes + 255) & ~(size_t)255; return p;
    };
    __bf16* hA     = (__bf16*)take((size_t)N_PAD * 256 * 2);
    __bf16* hB     = (__bf16*)take((size_t)N_PAD * 256 * 2);
    float*  hW     = (float*) take((size_t)N_PAD * 128 * 4);
    float*  accum  = (float*) take((size_t)N_PAD * 128 * 4);
    float*  deg    = (float*) take((size_t)N_PAD * 4);
    float*  dinv   = (float*) take((size_t)N_PAD * 4);
    float*  selfn  = (float*) take((size_t)N_PAD * 4);
    float*  norm   = (float*) take((size_t)EE * 4);
    __bf16* packed = (__bf16*)take((size_t)100000 * 2);

    // ---- pack all 10 weight matrices into WMMA B-fragment order ----
    const int FI[10]  = {256,128,128, 64, 32, 64,128, 64, 32, 64};
    const int FOP[10] = {128,128, 64, 32, 64,128, 64, 32, 64, 16}; // padded to x16
    size_t poff[10]; { size_t a = 0; for (int i = 0; i < 10; ++i) { poff[i] = a; a += (size_t)FI[i]*FOP[i]; } }
    for (int i = 0; i < 10; ++i) {
        const float* W = (i < 8) ? convW[i] : (i == 8 ? l0W : l1W);
        int fo_real = (i < 9) ? FOP[i] : 10;
        int total = FI[i] * FOP[i];
        pack_w<<<(total + 255) / 256, 256, 0, stream>>>(W, packed + poff[i], FI[i], fo_real, FOP[i]);
    }

    // ---- zero pad rows of the bf16 ping-pong buffers (rows NN..N_PAD) ----
    hipMemsetAsync(hA + (size_t)NN * 256, 0, (size_t)(N_PAD - NN) * 256 * 2, stream);
    hipMemsetAsync(hB + (size_t)NN * 256, 0, (size_t)(N_PAD - NN) * 256 * 2, stream);

    // ---- graph normalization ----
    deg_init<<<(NN + 255) / 256, 256, 0, stream>>>(deg);
    deg_acc <<<(EE + 255) / 256, 256, 0, stream>>>(colp, ew, deg);
    dinv_k  <<<(NN + 255) / 256, 256, 0, stream>>>(deg, dinv, selfn);
    norm_k  <<<(EE + 255) / 256, 256, 0, stream>>>(rowp, colp, ew, dinv, norm);

    // ---- input BN -> hA ----
    bn_input<<<(NN * 256 + 255) / 256, 256, 0, stream>>>(x, bnG[0], bnB[0], bnM[0], bnV[0], hA);

    // ---- 8 GCN conv layers ----
    const int FO[8]     = {128, 128, 64, 32, 64, 128, 64, 32};
    const int FOSH[8]   = {  7,   7,  6,  5,  6,   7,  6,  5};
    __bf16* bufs[2] = {hA, hB};
    for (int i = 0; i < 8; ++i) {
        __bf16* in  = bufs[i & 1];
        __bf16* nxt = bufs[(i + 1) & 1];
        int nt = FO[i] / 16, ktc = FI[i] / 32;
        int groups = MG * ((nt + 1) / 2);
        gemm_bf16_wmma<<<(groups + 3) / 4, 128, 0, stream>>>(in, packed + poff[i], hW, FO[i], nt, ktc, groups);
        hipMemsetAsync(accum, 0, (size_t)NN * FO[i] * 4, stream);
        {
            int fo4_shift = FOSH[i] - 2;
            long long total = (long long)EE << fo4_shift;
            scatter_edges<<<(unsigned)((total + 255) / 256), 256, 0, stream>>>(
                hW, rowp, colp, norm, accum, fo4_shift);
        }
        // ReLU everywhere: layers 0..6 per reference loop; layer 7's output is
        // only ever consumed as relu(h) by lin0.
        finalize_conv<<<((NN << FOSH[i]) + 255) / 256, 256, 0, stream>>>(
            accum, hW, selfn, convB[i], bnG[i + 1], bnB[i + 1], bnM[i + 1], bnV[i + 1],
            nxt, FOSH[i]);
    }

    // ---- lin0: [N,32] @ [32,64], BN9, relu ----  (input is bufs[0] = hA)
    {
        int nt = 4, ktc = 1, groups = MG * 2;
        gemm_bf16_wmma<<<(groups + 3) / 4, 128, 0, stream>>>(hA, packed + poff[8], hW, 64, nt, ktc, groups);
        finalize_lin<<<(NN * 64 + 255) / 256, 256, 0, stream>>>(hW, l0b, bnG[9], bnB[9], bnM[9], bnV[9], hB);
    }
    // ---- lin1: [N,64] @ [64,10 (pad 16)] + bias -> d_out ----
    {
        int nt = 1, ktc = 2, groups = MG;
        gemm_bf16_wmma<<<(groups + 3) / 4, 128, 0, stream>>>(hB, packed + poff[9], hW, 16, nt, ktc, groups);
        write_out<<<(NN * 10 + 255) / 256, 256, 0, stream>>>(hW, l1b, out);
    }
}
